// AgentAttention_26190710571407
// MI455X (gfx1250) — matrix operand order repaired and checked by reading, not compiled, not run on hardware
//
#include <hip/hip_runtime.h>
#include <hip/hip_bf16.h>
#include <math.h>

// ---------------------------------------------------------------------------
// Agent attention for MI455X (gfx1250, wave32, WMMA).
// All matmuls use v_wmma_f32_16x16x32_bf16. Softmax stats are f32 with
// 16-lane __shfl_xor reductions matching the 16x16 C/D fragment layout.
// This revision: native __bf16 casts (packed v_cvt instead of bit-twiddled
// VALU), LDS staging of the stage-1 V chunk (vmem->ds), global prefetch.
// Workspace usage: ~99 MB (kA, vA, ctx2 in bf16 + small buffers).
// ---------------------------------------------------------------------------

#define BATCH 4
#define NH    16
#define NTOK  4096
#define HD    64
#define DM    1024

typedef __bf16 bf16;
typedef __bf16 v16bf __attribute__((ext_vector_type(16)));
typedef __bf16 v8bf  __attribute__((ext_vector_type(8)));
typedef float  v8f   __attribute__((ext_vector_type(8)));

union BF16Frag { v16bf v; bf16 e[16]; };

__device__ __forceinline__ v8f v8f_splat(float x) {
  v8f r;
#pragma unroll
  for (int i = 0; i < 8; ++i) r[i] = x;
  return r;
}

__device__ __forceinline__ v8f wmma_bf(v16bf a, v16bf b, v8f c) {
  // (neg_a, A, neg_b, B, c_mod, C, reuse_a, reuse_b)
  return __builtin_amdgcn_wmma_f32_16x16x32_bf16(false, a, false, b, (short)0, c,
                                                 false, false);
}

// ---- Fragment loaders -------------------------------------------------------
// A matrix 16x32 bf16 (ISA 7.12.2): lane = row M (mod 16), half = lane>>4,
// VGPR v<4 -> K = 2v + 8*half ; v>=4 -> K = 2v + 8 + 8*half (pairs contiguous).
__device__ __forceinline__ v16bf loadA_f32(const float* p, int ld) {
  int lane = threadIdx.x & 31, half = lane >> 4, r = lane & 15;
  const float* row = p + (size_t)r * ld;
  BF16Frag a;
#pragma unroll
  for (int v = 0; v < 8; ++v) {
    int k = 2 * (v & 3) + ((v & 4) ? 16 : 0) + 8 * half;
    a.e[2 * v]     = (bf16)row[k];
    a.e[2 * v + 1] = (bf16)row[k + 1];
  }
  return a.v;
}

__device__ __forceinline__ v16bf loadA_bf(const bf16* p, int ld) {
  int lane = threadIdx.x & 31, half = lane >> 4, r = lane & 15;
  const bf16* row = p + (size_t)r * ld;
  BF16Frag a;
#pragma unroll
  for (int v = 0; v < 8; ++v) {
    int k = 2 * (v & 3) + ((v & 4) ? 16 : 0) + 8 * half;
    a.e[2 * v]     = row[k];
    a.e[2 * v + 1] = row[k + 1];
  }
  return a.v;
}

// B matrix 32x16 bf16: lane = col N (mod 16); same K order as A fragment.
// "Transposed source": element (k,n) = W[n*ld + k]  (weight rows, contiguous K)
__device__ __forceinline__ v16bf loadBT_bf(const bf16* W, int ld) {
  int lane = threadIdx.x & 31, half = lane >> 4, n = lane & 15;
  const bf16* row = W + (size_t)n * ld;
  BF16Frag b;
#pragma unroll
  for (int v = 0; v < 8; ++v) {
    int k = 2 * (v & 3) + ((v & 4) ? 16 : 0) + 8 * half;
    b.e[2 * v]     = row[k];
    b.e[2 * v + 1] = row[k + 1];
  }
  return b.v;
}

// Direct source: element (k,n) = S[k*ld + n]  (row-major [K][N], strided K).
// Use only on LDS-resident tiles (bank-parallel ds reads).
__device__ __forceinline__ v16bf loadB_bf(const bf16* S, int ld) {
  int lane = threadIdx.x & 31, half = lane >> 4, n = lane & 15;
  BF16Frag b;
#pragma unroll
  for (int v = 0; v < 8; ++v) {
    int k = 2 * (v & 3) + ((v & 4) ? 16 : 0) + 8 * half;
    b.e[2 * v]     = S[(size_t)k * ld + n];
    b.e[2 * v + 1] = S[(size_t)(k + 1) * ld + n];
  }
  return b.v;
}

// C/D 16x16 f32: VGPR r -> row M = r + 8*half, col = lane & 15.
__device__ __forceinline__ void storeD_f32(float* p, int ld, v8f d) {
  int lane = threadIdx.x & 31, half = lane >> 4, c = lane & 15;
#pragma unroll
  for (int r = 0; r < 8; ++r) p[(size_t)(r + 8 * half) * ld + c] = d[r];
}

__device__ __forceinline__ void storeD_bf(bf16* p, int ld, v8f d) {
  int lane = threadIdx.x & 31, half = lane >> 4, c = lane & 15;
#pragma unroll
  for (int r = 0; r < 8; ++r) p[(size_t)(r + 8 * half) * ld + c] = (bf16)d[r];
}

// ---- Kernel 0: f32 -> bf16 conversion (weights) -----------------------------
__global__ void k_cvt_bf16(const float* __restrict__ src,
                           bf16* __restrict__ dst, int n) {
  for (int i = blockIdx.x * blockDim.x + threadIdx.x; i < n;
       i += gridDim.x * blockDim.x)
    dst[i] = (bf16)src[i];
}

// ---- Shared tile: y[16x64] = x[16x64] @ W^T + b, bf16 out -------------------
__device__ __forceinline__ void proj_tile16(const float* src,
                                            const bf16* Wh,
                                            const float* bias,
                                            bf16* dst) {
  v16bf a0 = loadA_f32(src, HD);
  v16bf a1 = loadA_f32(src + 32, HD);
  int lane = threadIdx.x & 31;
#pragma unroll
  for (int t = 0; t < 4; ++t) {
    v8f acc = v8f_splat(bias[t * 16 + (lane & 15)]);
    acc = wmma_bf(a0, loadBT_bf(Wh + (size_t)(t * 16) * HD, HD), acc);
    acc = wmma_bf(a1, loadBT_bf(Wh + (size_t)(t * 16) * HD + 32, HD), acc);
    storeD_bf(dst + t * 16, HD, acc);
  }
}

// ---- Kernel 1: q_agents[h] = agents[h] @ Wq^T + bq (per head, 4 waves) ------
__global__ void k_qagents(const float* __restrict__ ag,
                          const bf16* __restrict__ WqH,
                          const float* __restrict__ bq,
                          bf16* __restrict__ qA) {
  int h = blockIdx.x, wave = threadIdx.x >> 5;
  size_t off = (size_t)h * 64 * HD + (size_t)wave * 16 * HD;
  proj_tile16(ag + off, WqH, bq, qA + off);
}

// ---- Kernel 2: k_agents / v_agents projections ------------------------------
__global__ void k_proj_kv(const float* __restrict__ kin,
                          const float* __restrict__ vin,
                          const bf16* __restrict__ WkH,
                          const float* __restrict__ bk,
                          const bf16* __restrict__ WvH,
                          const float* __restrict__ bv,
                          bf16* __restrict__ kA,
                          bf16* __restrict__ vA) {
  int wave = threadIdx.x >> 5;                       // 8 waves x 16 rows
  int nt = blockIdx.x % (NTOK / 128);
  int bh = blockIdx.x / (NTOK / 128);
  size_t off = ((size_t)bh * NTOK + (size_t)nt * 128 + wave * 16) * HD;
  proj_tile16(kin + off, WkH, bk, kA + off);
  proj_tile16(vin + off, WvH, bv, vA + off);
}

// ---- Kernel 3: stage-1 agent attention (flash-style online softmax) ---------
// One block per (b,h); 4 waves, each owns 16 agent rows across all N keys.
// The 32x64 V chunk is cooperatively staged in LDS (16B vector copies) since
// all 4 waves consume the same chunk; B-fragments then come from ds reads.
__global__ void k_stage1(const bf16* __restrict__ qA,
                         const bf16* __restrict__ kA,
                         const bf16* __restrict__ vA,
                         bf16* __restrict__ ctx1) {
  __shared__ bf16 Pbuf[4][16 * 32];   // 4 KB, wave-private strips
  __shared__ bf16 Vs[32 * 64];        // 4 KB, shared V chunk
  int bh = blockIdx.x, h = bh & (NH - 1);
  int wave = threadIdx.x >> 5, lane = threadIdx.x & 31, half = lane >> 4;
  const bf16* qAw = qA + (size_t)h * 64 * HD + (size_t)wave * 16 * HD;
  const bf16* kAb = kA + (size_t)bh * NTOK * HD;
  const bf16* vAb = vA + (size_t)bh * NTOK * HD;
  v16bf aq0 = loadA_bf(qAw, HD);
  v16bf aq1 = loadA_bf(qAw + 32, HD);
  float m[8], l[8];
  v8f ctx[4];
#pragma unroll
  for (int r = 0; r < 8; ++r) { m[r] = -3.0e38f; l[r] = 0.f; }
#pragma unroll
  for (int t = 0; t < 4; ++t) ctx[t] = v8f_splat(0.f);
  const float scale = 0.125f;  // 1/sqrt(64)
  bf16* Ps = Pbuf[wave];

  for (int kb = 0; kb < NTOK; kb += 32) {
    const bf16* kp = kAb + (size_t)kb * HD;
    __builtin_prefetch(kAb + (size_t)(kb + 32) * HD, 0, 0);  // next K chunk
    v8f S0 = v8f_splat(0.f), S1 = v8f_splat(0.f);
    S0 = wmma_bf(aq0, loadBT_bf(kp, HD), S0);
    S0 = wmma_bf(aq1, loadBT_bf(kp + 32, HD), S0);
    S1 = wmma_bf(aq0, loadBT_bf(kp + 16 * HD, HD), S1);
    S1 = wmma_bf(aq1, loadBT_bf(kp + 16 * HD + 32, HD), S1);
#pragma unroll
    for (int r = 0; r < 8; ++r) {
      float s0 = S0[r] * scale, s1 = S1[r] * scale;
      float mx = fmaxf(s0, s1);
#pragma unroll
      for (int o = 8; o; o >>= 1) mx = fmaxf(mx, __shfl_xor(mx, o, 16));
      float mn = fmaxf(m[r], mx);
      float rs = __expf(m[r] - mn);
      float p0 = __expf(s0 - mn), p1 = __expf(s1 - mn);
      float sum = p0 + p1;
#pragma unroll
      for (int o = 8; o; o >>= 1) sum += __shfl_xor(sum, o, 16);
      l[r] = l[r] * rs + sum;
      m[r] = mn;
#pragma unroll
      for (int t = 0; t < 4; ++t) ctx[t][r] *= rs;
      S0[r] = p0; S1[r] = p1;
    }
    __syncthreads();                    // protect LDS reuse (P strips + Vs)
    // stage V chunk: 32 rows x 64 cols bf16 = 2048 elems; 128 thr x 2 x v8bf
    {
      const v8bf* vsrc = (const v8bf*)(vAb + (size_t)kb * HD);
      v8bf* vdst = (v8bf*)Vs;
      int tid = threadIdx.x;
      vdst[tid]       = vsrc[tid];
      vdst[tid + 128] = vsrc[tid + 128];
    }
#pragma unroll
    for (int r = 0; r < 8; ++r) {
      Ps[(r + 8 * half) * 32 + (lane & 15)]      = (bf16)S0[r];
      Ps[(r + 8 * half) * 32 + 16 + (lane & 15)] = (bf16)S1[r];
    }
    __syncthreads();
    v16bf ap = loadA_bf(Ps, 32);        // P chunk as A-fragment (K=32 keys)
#pragma unroll
    for (int t = 0; t < 4; ++t)
      ctx[t] = wmma_bf(ap, loadB_bf(Vs + t * 16, HD), ctx[t]);
  }
  bf16* outp = ctx1 + (size_t)bh * 64 * HD + (size_t)wave * 16 * HD;
#pragma unroll
  for (int t = 0; t < 4; ++t)
#pragma unroll
    for (int r = 0; r < 8; ++r)
      outp[(size_t)(r + 8 * half) * HD + t * 16 + (lane & 15)] =
          (bf16)(ctx[t][r] / l[r]);
}

// ---- Kernel 4: stage-2 (queries attend to agent context) --------------------
// 8 waves x 16 query rows; full 64-agent row in register -> exact softmax.
// Emits scores2 (output #2) and context2 in [B,N,1024] bf16 (head transposed).
__global__ void k_stage2(const float* __restrict__ q,
                         const bf16* __restrict__ ctx1,
                         float* __restrict__ sc2,
                         bf16* __restrict__ ctx2) {
  __shared__ bf16 c1s[64 * HD];       // 8 KB
  __shared__ bf16 Pbuf[8][16 * 64];   // 16 KB
  int nt = blockIdx.x & 31;           // NTOK/128 = 32 tiles
  int bh = blockIdx.x >> 5;
  int b = bh >> 4, h = bh & 15;
  int wave = threadIdx.x >> 5, lane = threadIdx.x & 31, half = lane >> 4;
  const bf16* c1g = ctx1 + (size_t)bh * 64 * HD;
  for (int i = threadIdx.x; i < (64 * HD) / 8; i += 256)
    ((v8bf*)c1s)[i] = ((const v8bf*)c1g)[i];
  __syncthreads();
  int row0 = nt * 128 + wave * 16;
  const float* qrow = q + ((size_t)bh * NTOK + row0) * HD;
  v16bf a0 = loadA_f32(qrow, HD);
  v16bf a1 = loadA_f32(qrow + 32, HD);
  const float scale = 0.125f;
  v8f S[4];
#pragma unroll
  for (int t = 0; t < 4; ++t) {                 // S2 = q @ ctx1^T
    S[t] = v8f_splat(0.f);
    S[t] = wmma_bf(a0, loadBT_bf(c1s + (size_t)(t * 16) * HD, HD), S[t]);
    S[t] = wmma_bf(a1, loadBT_bf(c1s + (size_t)(t * 16) * HD + 32, HD), S[t]);
  }
  bf16* Ps = Pbuf[wave];
  float* sc2row = sc2 + ((size_t)bh * NTOK + row0) * 64;
#pragma unroll
  for (int r = 0; r < 8; ++r) {
    float sv[4];
#pragma unroll
    for (int t = 0; t < 4; ++t) sv[t] = S[t][r] * scale;
    float mx = fmaxf(fmaxf(sv[0], sv[1]), fmaxf(sv[2], sv[3]));
#pragma unroll
    for (int o = 8; o; o >>= 1) mx = fmaxf(mx, __shfl_xor(mx, o, 16));
    float sum = 0.f;
#pragma unroll
    for (int t = 0; t < 4; ++t) { sv[t] = __expf(sv[t] - mx); sum += sv[t]; }
#pragma unroll
    for (int o = 8; o; o >>= 1) sum += __shfl_xor(sum, o, 16);
    float inv = 1.f / sum;
    int rr = r + 8 * half;
#pragma unroll
    for (int t = 0; t < 4; ++t) {
      float p = sv[t] * inv;
      sc2row[(size_t)rr * 64 + t * 16 + (lane & 15)] = p;   // output #2
      Ps[rr * 64 + t * 16 + (lane & 15)] = (bf16)p;
    }
  }
  __syncthreads();
  v16bf ap0 = loadA_bf(Ps, 64);
  v16bf ap1 = loadA_bf(Ps + 32, 64);
  bf16* outp = ctx2 + ((size_t)b * NTOK + row0) * DM + h * HD;
#pragma unroll
  for (int t = 0; t < 4; ++t) {                 // ctx2 = P2 @ ctx1
    v8f acc = v8f_splat(0.f);
    acc = wmma_bf(ap0, loadB_bf(c1s + t * 16, HD), acc);
    acc = wmma_bf(ap1, loadB_bf(c1s + 32 * HD + t * 16, HD), acc);
    storeD_bf(outp + t * 16, DM, acc);
  }
}

// ---- Kernel 5: output projection out = ctx2 @ Wo^T + bo ---------------------
// Grid (128 row-blocks, 16 col-blocks); 8 waves x 16 rows, K = 1024.
__global__ void k_outproj(const bf16* __restrict__ x,
                          const bf16* __restrict__ WoH,
                          const float* __restrict__ bo,
                          float* __restrict__ out) {
  int wave = threadIdx.x >> 5, lane = threadIdx.x & 31;
  size_t row0 = (size_t)blockIdx.x * 128 + wave * 16;
  int col0 = blockIdx.y * 64;
  const bf16* xr = x + row0 * DM;
  v8f acc[4];
#pragma unroll
  for (int t = 0; t < 4; ++t)
    acc[t] = v8f_splat(bo[col0 + t * 16 + (lane & 15)]);
  for (int ks = 0; ks < DM; ks += 32) {
    __builtin_prefetch(xr + ks + 128, 0, 0);                   // next A strips
    __builtin_prefetch(WoH + (size_t)col0 * DM + ks + 128, 0, 3); // Wo stays hot
    v16bf a = loadA_bf(xr + ks, DM);
#pragma unroll
    for (int t = 0; t < 4; ++t)
      acc[t] = wmma_bf(a, loadBT_bf(WoH + (size_t)(col0 + t * 16) * DM + ks, DM),
                       acc[t]);
  }
#pragma unroll
  for (int t = 0; t < 4; ++t)
    storeD_f32(out + row0 * DM + col0 + t * 16, DM, acc[t]);
}

// ---------------------------------------------------------------------------
extern "C" void kernel_launch(void* const* d_in, const int* in_sizes, int n_in,
                              void* d_out, int out_size, void* d_ws,
                              size_t ws_size, hipStream_t stream) {
  (void)in_sizes; (void)n_in; (void)out_size; (void)ws_size;
  const float* q  = (const float*)d_in[0];
  const float* k  = (const float*)d_in[1];
  const float* v  = (const float*)d_in[2];
  const float* Wq = (const float*)d_in[3];
  const float* bq = (const float*)d_in[4];
  const float* Wk = (const float*)d_in[5];
  const float* bk = (const float*)d_in[6];
  const float* Wv = (const float*)d_in[7];
  const float* bv = (const float*)d_in[8];
  const float* Wo = (const float*)d_in[9];
  const float* bo = (const float*)d_in[10];
  const float* ag = (const float*)d_in[11];

  float* out     = (float*)d_out;
  float* scores2 = out + (size_t)BATCH * NTOK * DM;

  // Workspace layout (bytes), total ~99 MiB
  char* ws = (char*)d_ws;
  bf16* qA  = (bf16*)(ws);                                            // 128 KB
  bf16* kA  = (bf16*)(ws + 131072);                                   // 32 MB
  bf16* vA  = (bf16*)(ws + 131072 + 33554432);                        // 32 MB
  bf16* c1h = (bf16*)(ws + 131072 + 2u * 33554432u);                  // 512 KB
  bf16* c2  = (bf16*)(ws + 131072 + 2u * 33554432u + 524288u);        // 32 MB
  bf16* WoH = (bf16*)(ws + 131072 + 3ull * 33554432ull + 524288u);    // 2 MB
  bf16* WqH = (bf16*)((char*)WoH + 2097152);
  bf16* WkH = (bf16*)((char*)WqH + 8192);
  bf16* WvH = (bf16*)((char*)WkH + 8192);

  // 0) weight conversions (tiny, L2-resident afterwards)
  k_cvt_bf16<<<2048, 256, 0, stream>>>(Wo, WoH, DM * DM);
  k_cvt_bf16<<<16, 256, 0, stream>>>(Wq, WqH, HD * HD);
  k_cvt_bf16<<<16, 256, 0, stream>>>(Wk, WkH, HD * HD);
  k_cvt_bf16<<<16, 256, 0, stream>>>(Wv, WvH, HD * HD);
  // 1) q_agents (per head)
  k_qagents<<<NH, 128, 0, stream>>>(ag, WqH, bq, qA);
  // 2) k/v projections
  k_proj_kv<<<BATCH * NH * (NTOK / 128), 256, 0, stream>>>(k, v, WkH, bk, WvH,
                                                           bv, kA, vA);
  // 3) stage-1: agents attend to keys/values (online softmax over N)
  k_stage1<<<BATCH * NH, 128, 0, stream>>>(qA, kA, vA, c1h);
  // 4) stage-2: queries attend to agent context; emits scores2 + ctx2
  k_stage2<<<BATCH * NH * (NTOK / 128), 256, 0, stream>>>(q, c1h, scores2, c2);
  // 5) output projection
  k_outproj<<<dim3(128, 16), 256, 0, stream>>>(c2, WoH, bo, out);
}
